// AttModule_mamba_4088808865958
// MI455X (gfx1250) — compile-verified, matrix operand order checked by reading, not run on hardware
//
#include <hip/hip_runtime.h>

// ---- problem constants (match reference) ----
#define B_   2
#define L_   2048
#define C_   512
#define DI_  1024
#define N_   16
#define R_   32
#define K_   7
#define DC_  4
#define LP_  (L_ + 6)     // padded length for K=7 same-conv (3 each side)
#define KC_  32           // K-chunk per LDS panel step

typedef unsigned short u16;
typedef __attribute__((ext_vector_type(16))) __bf16 v16bf;
typedef __attribute__((ext_vector_type(8)))  __bf16 v8bf;
typedef __attribute__((ext_vector_type(8)))  float  v8f;

// ---------- helpers ----------
__device__ __forceinline__ u16 f2bf_bits(float f) {
  unsigned int u = __builtin_bit_cast(unsigned int, f);
  u += 0x7fffu + ((u >> 16) & 1u);   // round-to-nearest-even
  return (u16)(u >> 16);
}
__device__ __forceinline__ __bf16 bits2bf(u16 b) {
  return __builtin_bit_cast(__bf16, b);
}
__device__ __forceinline__ float silu_f(float v) {
  return v / (1.f + __expf(-v));
}
// 16-bit fragment layout: lane (r, h) holds K = h*8+[0,8) then 16+h*8+[0,8),
// both contiguous -> two 16B vector loads + shuffle (folds into load layout).
__device__ __forceinline__ v16bf load_frag(const u16* rowbase, int k0, int h) {
  const v8bf* q = (const v8bf*)(rowbase + k0 + h * 8);
  v8bf lo = q[0];          // K = k0 + h*8 + [0,8)
  v8bf hi = q[2];          // K = k0 + 16 + h*8 + [0,8)
  return __builtin_shufflevector(lo, hi, 0,1,2,3,4,5,6,7,8,9,10,11,12,13,14,15);
}
__device__ __forceinline__ int kmap(int i, int h) {
  return (i < 8) ? (h * 8 + i) : (8 + h * 8 + i);
}
// ---- CDNA5 async copy global->LDS (ASYNCcnt-tracked), 16B per lane ----
__device__ __forceinline__ void async_copy16(unsigned lds_off, const void* gaddr) {
  asm volatile("global_load_async_to_lds_b128 %0, %1, off"
               :: "v"(lds_off), "v"(gaddr) : "memory");
}
__device__ __forceinline__ void wait_async0() {
  asm volatile("s_wait_asynccnt 0x0" ::: "memory");
}

// ---------- conversion / packing kernels ----------
__global__ void cvt_f32_bf16_kernel(const float* __restrict__ src,
                                    u16* __restrict__ dst, int n) {
  int i = blockIdx.x * blockDim.x + threadIdx.x;
  if (i < n) dst[i] = f2bf_bits(src[i]);
}

// src (rows x cols) row-major f32 -> dst (cols x rows) row-major bf16
__global__ void cvt_transpose_kernel(const float* __restrict__ src,
                                     u16* __restrict__ dst, int rows, int cols) {
  int i = blockIdx.x * blockDim.x + threadIdx.x;
  if (i >= rows * cols) return;
  int r = i / cols, c = i % cols;
  dst[c * rows + r] = f2bf_bits(src[i]);
}

// ff_w (m, c, k) f32 -> ffwT (k, m, c) bf16  [contiguous in c for A fragments]
__global__ void pack_ffw_kernel(const float* __restrict__ src, u16* __restrict__ dst) {
  int i = blockIdx.x * blockDim.x + threadIdx.x;
  if (i >= C_ * C_ * K_) return;
  int m = i / (C_ * K_);
  int rem = i % (C_ * K_);
  int c = rem / K_, k = rem % K_;
  dst[(k * C_ + m) * C_ + c] = f2bf_bits(src[i]);
}

// x (b, c, l) f32 -> xpadT (b, lp, c) bf16 with lp in [0, L+6), zero borders
__global__ void pack_xpad_kernel(const float* __restrict__ x, u16* __restrict__ xpadT) {
  int i = blockIdx.x * blockDim.x + threadIdx.x;
  if (i >= B_ * LP_ * C_) return;
  int c  = i & (C_ - 1);
  int lp = (i / C_) % LP_;
  int b  = i / (C_ * LP_);
  int l  = lp - 3;
  float v = (l >= 0 && l < L_) ? x[(b * C_ + c) * L_ + l] : 0.f;
  xpadT[i] = f2bf_bits(v);
}

// ---------- 1) ff conv (K=7) as implicit WMMA GEMM, 16(M) x 64(L) per wave ----------
// ffwT: (k, Cout, Cin) bf16 ; xpadT: (B, L+6, Cin) bf16 ; convout: (B, Cout, L) f32
__global__ void conv_ffn_wmma_kernel(const u16* __restrict__ ffwT,
                                     const u16* __restrict__ xpadT,
                                     const float* __restrict__ ffb,
                                     float* __restrict__ convout) {
  int wid = blockIdx.x * 8 + (threadIdx.x >> 5);  // 2048 waves
  int b   = wid >> 10;                            // 1024 tiles per batch
  int rem = wid & 1023;
  int mt  = rem >> 5;                             // 32 Cout tiles
  int g   = rem & 31;                             // 32 groups of 64 L-cols
  int lane = threadIdx.x & 31;
  int r = lane & 15, h = lane >> 4;
  int row = mt * 16 + r;

  v8f acc0 = {0,0,0,0,0,0,0,0}, acc1 = acc0, acc2 = acc0, acc3 = acc0;
  for (int k = 0; k < K_; ++k) {
    const u16* arow = ffwT + (size_t)(k * C_ + row) * C_;
    // output col l -> padded index lp = l + k ; branch-free via zero padding
    const u16* bbase = xpadT + (size_t)(b * LP_ + g * 64 + r + k) * C_;
    for (int k0 = 0; k0 < C_; k0 += 32) {
      v16bf a = load_frag(arow, k0, h);
      __builtin_prefetch(arow + k0 + 32, 0, 1);
      v16bf b0 = load_frag(bbase +  0 * 16 * C_, k0, h);
      v16bf b1 = load_frag(bbase +  1 * 16 * C_, k0, h);
      v16bf b2 = load_frag(bbase +  2 * 16 * C_, k0, h);
      v16bf b3 = load_frag(bbase +  3 * 16 * C_, k0, h);
      acc0 = __builtin_amdgcn_wmma_f32_16x16x32_bf16(false, a, false, b0, (short)0, acc0, false, false);
      acc1 = __builtin_amdgcn_wmma_f32_16x16x32_bf16(false, a, false, b1, (short)0, acc1, false, false);
      acc2 = __builtin_amdgcn_wmma_f32_16x16x32_bf16(false, a, false, b2, (short)0, acc2, false, false);
      acc3 = __builtin_amdgcn_wmma_f32_16x16x32_bf16(false, a, false, b3, (short)0, acc3, false, false);
    }
  }
  v8f accs[4] = {acc0, acc1, acc2, acc3};
#pragma unroll
  for (int t = 0; t < 4; ++t) {
    int ocol = g * 64 + t * 16 + r;
#pragma unroll
    for (int j = 0; j < 8; ++j) {
      int orow = mt * 16 + j + 8 * h;
      float v = accs[t][j] + ffb[orow];
      convout[(b * C_ + orow) * L_ + ocol] = fmaxf(v, 0.f);
    }
  }
}

// ---------- 2) BN statistics per channel over (B, L) ----------
__global__ void bn_stats_kernel(const float* __restrict__ convout,
                                float* __restrict__ mean,
                                float* __restrict__ rstd) {
  __shared__ float ss[256], sq[256];
  int c = blockIdx.x;
  float s = 0.f, q = 0.f;
  for (int i = threadIdx.x; i < B_ * L_; i += 256) {
    int b = i >> 11, l = i & (L_ - 1);
    float v = convout[(b * C_ + c) * L_ + l];
    s += v; q += v * v;
  }
  ss[threadIdx.x] = s; sq[threadIdx.x] = q;
  __syncthreads();
  for (int st = 128; st > 0; st >>= 1) {
    if (threadIdx.x < st) {
      ss[threadIdx.x] += ss[threadIdx.x + st];
      sq[threadIdx.x] += sq[threadIdx.x + st];
    }
    __syncthreads();
  }
  if (threadIdx.x == 0) {
    float mu  = ss[0] * (1.f / (B_ * L_));
    float var = sq[0] * (1.f / (B_ * L_)) - mu * mu;
    mean[c] = mu;
    rstd[c] = rsqrtf(var + 1e-5f);
  }
}

// ---------- 3) normalize + mask + transpose to (B*L, C) bf16 ----------
__global__ void bn_norm_transpose_kernel(const float* __restrict__ convout,
                                         const float* __restrict__ mean,
                                         const float* __restrict__ rstd,
                                         const float* __restrict__ gamma,
                                         const float* __restrict__ beta,
                                         const float* __restrict__ mask,
                                         u16* __restrict__ ubT) {
  int i = blockIdx.x * 256 + threadIdx.x;       // over B*C*L = 2^21
  int l = i & (L_ - 1);
  int c = (i >> 11) & (C_ - 1);
  int b = i >> 20;
  float v = (convout[i] - mean[c]) * rstd[c] * gamma[c] + beta[c];
  v *= mask[b * L_ + l];
  ubT[(b * L_ + l) * C_ + c] = f2bf_bits(v);
}

// ---------- generic bf16 WMMA GEMM with async-LDS B panel ----------
// D(MxN) = A(MxK) * BT(NxK)^T. Block = 128(M) x 64(N), 8 waves; B panel shared
// via double-buffered LDS filled with GLOBAL_LOAD_ASYNC_TO_LDS_B128 (ASYNCcnt).
__global__ void gemm_bf16_wmma_kernel(const u16* __restrict__ A,
                                      const u16* __restrict__ BT,
                                      float* __restrict__ D,
                                      int M, int N, int Kd) {
  __shared__ u16 panel[2][64 * KC_];            // 2 x 4KB
  int nblocks = N >> 6;
  int mb = blockIdx.x / nblocks;                // M block of 128 rows
  int g  = blockIdx.x % nblocks;                // N block of 64 cols
  int wave = threadIdx.x >> 5;
  int lane = threadIdx.x & 31;
  int r = lane & 15, h = lane >> 4;
  const u16* arow = A + (size_t)(mb * 128 + wave * 16 + r) * Kd;

  // cooperative panel fill: thread t copies 16B: row n = t>>2, k-part = (t&3)*8
  int tn = threadIdx.x >> 2;
  int tk = (threadIdx.x & 3) * 8;
  const u16* gsrc = BT + (size_t)(g * 64 + tn) * Kd + tk;
  unsigned lds0 = (unsigned)(size_t)&panel[0][tn * KC_ + tk];
  unsigned lds1 = (unsigned)(size_t)&panel[1][tn * KC_ + tk];

  v8f acc0 = {0,0,0,0,0,0,0,0}, acc1 = acc0, acc2 = acc0, acc3 = acc0;

  async_copy16(lds0, gsrc);                     // prologue: panel 0 (k0 = 0)
  int nsteps = Kd / KC_;
  for (int s = 0; s < nsteps; ++s) {
    wait_async0();                              // my slice landed
    __syncthreads();                            // everyone's slice landed
    if (s + 1 < nsteps)                         // prefetch next panel (other buffer)
      async_copy16((s & 1) ? lds0 : lds1, gsrc + (s + 1) * KC_);
    const u16* bp = &panel[s & 1][0];
    int k0 = s * KC_;
    v16bf a = load_frag(arow, k0, h);
    __builtin_prefetch(arow + k0 + KC_, 0, 1);
    v16bf b0 = load_frag(bp + (0 * 16 + r) * KC_, 0, h);
    v16bf b1 = load_frag(bp + (1 * 16 + r) * KC_, 0, h);
    v16bf b2 = load_frag(bp + (2 * 16 + r) * KC_, 0, h);
    v16bf b3 = load_frag(bp + (3 * 16 + r) * KC_, 0, h);
    acc0 = __builtin_amdgcn_wmma_f32_16x16x32_bf16(false, a, false, b0, (short)0, acc0, false, false);
    acc1 = __builtin_amdgcn_wmma_f32_16x16x32_bf16(false, a, false, b1, (short)0, acc1, false, false);
    acc2 = __builtin_amdgcn_wmma_f32_16x16x32_bf16(false, a, false, b2, (short)0, acc2, false, false);
    acc3 = __builtin_amdgcn_wmma_f32_16x16x32_bf16(false, a, false, b3, (short)0, acc3, false, false);
  }
  v8f accs[4] = {acc0, acc1, acc2, acc3};
#pragma unroll
  for (int t = 0; t < 4; ++t) {
    int col = g * 64 + t * 16 + r;
#pragma unroll
    for (int j = 0; j < 8; ++j) {
      long orow = mb * 128 + wave * 16 + j + 8 * h;
      D[orow * (long)N + col] = accs[t][j];
    }
  }
}

// ---------- 4) depthwise causal conv (DC=4) + bias + SiLU ----------
__global__ void dwconv_silu_kernel(const float* __restrict__ xz,
                                   const float* __restrict__ cw,
                                   const float* __restrict__ cb,
                                   float* __restrict__ xs_act,
                                   u16* __restrict__ xs_actb) {
  int i = blockIdx.x * 256 + threadIdx.x;       // over B*L*DI = 2^22
  int d = i & (DI_ - 1);
  int l = (i >> 10) & (L_ - 1);
  int b = i >> 21;
  float acc = cb[d];
#pragma unroll
  for (int j = 0; j < DC_; ++j) {
    int lp = l - (DC_ - 1) + j;
    if (lp >= 0)
      acc += cw[d * DC_ + j] * xz[((b * L_ + lp) * (2 * DI_)) + d];
  }
  float sv = silu_f(acc);
  int o = (b * L_ + l) * DI_ + d;
  xs_act[o]  = sv;
  xs_actb[o] = f2bf_bits(sv);
}

// ---------- 5) dt = softplus(dbc[:, :32] @ w_dt + dt_bias), 16x64 per wave ----------
// wdtT: (DI, 32) bf16 (transposed w_dt)
__global__ void gemm_dt_wmma_kernel(const float* __restrict__ dbc,
                                    const u16* __restrict__ wdtT,
                                    const float* __restrict__ dt_bias,
                                    float* __restrict__ dtb) {
  const int N = DI_;
  int wid = blockIdx.x * 8 + (threadIdx.x >> 5);  // 4096 waves
  int mt = wid >> 4;                              // 256 M tiles
  int g  = wid & 15;                              // 16 groups of 64 cols
  int lane = threadIdx.x & 31;
  int r = lane & 15, h = lane >> 4;
  const float* arow = dbc + (size_t)(mt * 16 + r) * 64;
  v16bf a;
#pragma unroll
  for (int i = 0; i < 16; ++i) a[i] = bits2bf(f2bf_bits(arow[kmap(i, h)]));
  v8f acc0 = {0,0,0,0,0,0,0,0}, acc1 = acc0, acc2 = acc0, acc3 = acc0;
  v16bf b0 = load_frag(wdtT + (size_t)(g * 64 +  0 + r) * R_, 0, h);
  v16bf b1 = load_frag(wdtT + (size_t)(g * 64 + 16 + r) * R_, 0, h);
  v16bf b2 = load_frag(wdtT + (size_t)(g * 64 + 32 + r) * R_, 0, h);
  v16bf b3 = load_frag(wdtT + (size_t)(g * 64 + 48 + r) * R_, 0, h);
  acc0 = __builtin_amdgcn_wmma_f32_16x16x32_bf16(false, a, false, b0, (short)0, acc0, false, false);
  acc1 = __builtin_amdgcn_wmma_f32_16x16x32_bf16(false, a, false, b1, (short)0, acc1, false, false);
  acc2 = __builtin_amdgcn_wmma_f32_16x16x32_bf16(false, a, false, b2, (short)0, acc2, false, false);
  acc3 = __builtin_amdgcn_wmma_f32_16x16x32_bf16(false, a, false, b3, (short)0, acc3, false, false);
  v8f accs[4] = {acc0, acc1, acc2, acc3};
#pragma unroll
  for (int t = 0; t < 4; ++t) {
    int col = g * 64 + t * 16 + r;
#pragma unroll
    for (int j = 0; j < 8; ++j) {
      int orow = mt * 16 + j + 8 * h;
      float v = accs[t][j] + dt_bias[col];
      v = (v > 20.f) ? v : log1pf(__expf(v));     // softplus
      dtb[orow * N + col] = v;
    }
  }
}

// ---------- 6) selective scan: thread = one (b, d); 16 states in registers ----------
__global__ void scan_kernel(const float* __restrict__ dtb,
                            const float* __restrict__ xs_act,
                            const float* __restrict__ dbc,
                            const float* __restrict__ xz,
                            const float* __restrict__ A_log,
                            const float* __restrict__ D_skip,
                            u16* __restrict__ ygb) {
  int b = blockIdx.x >> 2;                        // grid = B * (DI/256) = 8
  int d = (blockIdx.x & 3) * 256 + threadIdx.x;
  float Arow[N_], hst[N_];
#pragma unroll
  for (int n = 0; n < N_; ++n) {
    Arow[n] = -__expf(A_log[d * N_ + n]);
    hst[n] = 0.f;
  }
  float Dv = D_skip[d];
  for (int l = 0; l < L_; ++l) {
    long roff = (long)(b * L_ + l);
    float dt = dtb[roff * DI_ + d];
    float xv = xs_act[roff * DI_ + d];
    const float4* bc4 = (const float4*)(dbc + roff * 64 + R_);  // Bm[16], Cm[16]
    float bm[N_], cm[N_];
#pragma unroll
    for (int q = 0; q < 4; ++q) {
      float4 v = bc4[q];
      bm[q*4+0]=v.x; bm[q*4+1]=v.y; bm[q*4+2]=v.z; bm[q*4+3]=v.w;
      float4 w = bc4[4 + q];
      cm[q*4+0]=w.x; cm[q*4+1]=w.y; cm[q*4+2]=w.z; cm[q*4+3]=w.w;
    }
    float y = 0.f;
#pragma unroll
    for (int n = 0; n < N_; ++n) {
      float dA = __expf(dt * Arow[n]);
      hst[n] = dA * hst[n] + dt * bm[n] * xv;
      y += hst[n] * cm[n];
    }
    float z = xz[roff * (2 * DI_) + DI_ + d];
    float yg = (y + Dv * xv) * silu_f(z);
    ygb[roff * DI_ + d] = f2bf_bits(yg);
  }
}

// ---------- 7) epilogue: (x + conv_out + alpha*att) * mask, back to (B, C, L) ----------
__global__ void epilogue_kernel(const float* __restrict__ x,
                                const float* __restrict__ convout,
                                const float* __restrict__ attT,
                                const float* __restrict__ mask,
                                float* __restrict__ out) {
  int i = blockIdx.x * 256 + threadIdx.x;        // over B*C*L
  int l = i & (L_ - 1);
  int c = (i >> 11) & (C_ - 1);
  int b = i >> 20;
  float att = attT[(b * L_ + l) * C_ + c];
  out[i] = (x[i] + convout[i] + 1.0f * att) * mask[b * L_ + l];
}

// =====================================================================
extern "C" void kernel_launch(void* const* d_in, const int* in_sizes, int n_in,
                              void* d_out, int out_size, void* d_ws, size_t ws_size,
                              hipStream_t stream) {
  (void)in_sizes; (void)n_in; (void)out_size; (void)ws_size;
  const float* x       = (const float*)d_in[0];
  // d_in[1] ('f') unused by the reference
  const float* mask    = (const float*)d_in[2];
  const float* ff_w    = (const float*)d_in[3];
  const float* ff_b    = (const float*)d_in[4];
  const float* gamma   = (const float*)d_in[5];
  const float* beta    = (const float*)d_in[6];
  const float* w_in    = (const float*)d_in[7];
  const float* conv_w  = (const float*)d_in[8];
  const float* conv_b  = (const float*)d_in[9];
  const float* w_xproj = (const float*)d_in[10];
  const float* w_dt    = (const float*)d_in[11];
  const float* dt_bias = (const float*)d_in[12];
  const float* A_log   = (const float*)d_in[13];
  const float* D_skip  = (const float*)d_in[14];
  const float* w_out   = (const float*)d_in[15];
  float* out = (float*)d_out;

  // ---- workspace layout ----
  char* base = (char*)d_ws;
  size_t off = 0;
  auto wsalloc = [&](size_t bytes) -> void* {
    void* q = base + off;
    off = (off + bytes + 255) & ~(size_t)255;
    return q;
  };
  u16*   xpadT   = (u16*)  wsalloc((size_t)B_ * LP_ * C_ * 2);       // (B, L+6, C)
  u16*   ffwT    = (u16*)  wsalloc((size_t)K_ * C_ * C_ * 2);        // (k, m, c)
  u16*   winT    = (u16*)  wsalloc((size_t)(2 * DI_) * C_ * 2);      // (2DI, C)
  u16*   wxpT    = (u16*)  wsalloc((size_t)64 * DI_ * 2);            // (64, DI)
  u16*   wdtT    = (u16*)  wsalloc((size_t)DI_ * R_ * 2);            // (DI, 32)
  u16*   woutT   = (u16*)  wsalloc((size_t)C_ * DI_ * 2);            // (C, DI)
  float* convout = (float*)wsalloc((size_t)B_ * C_ * L_ * 4);
  float* bn_mean = (float*)wsalloc((size_t)C_ * 4);
  float* bn_rstd = (float*)wsalloc((size_t)C_ * 4);
  u16*   ubT     = (u16*)  wsalloc((size_t)B_ * L_ * C_ * 2);
  float* xz      = (float*)wsalloc((size_t)B_ * L_ * 2 * DI_ * 4);
  float* xs_act  = (float*)wsalloc((size_t)B_ * L_ * DI_ * 4);
  u16*   xs_actb = (u16*)  wsalloc((size_t)B_ * L_ * DI_ * 2);
  float* dbc     = (float*)wsalloc((size_t)B_ * L_ * 64 * 4);
  float* dtb     = (float*)wsalloc((size_t)B_ * L_ * DI_ * 4);
  u16*   ygb     = (u16*)  wsalloc((size_t)B_ * L_ * DI_ * 2);
  float* attT    = (float*)wsalloc((size_t)B_ * L_ * C_ * 4);

  // ---- pack weights/activations for contiguous WMMA fragments ----
  {
    int n = B_ * LP_ * C_;
    pack_xpad_kernel<<<(n + 255) / 256, 256, 0, stream>>>(x, xpadT);
  }
  {
    int n = C_ * C_ * K_;
    pack_ffw_kernel<<<(n + 255) / 256, 256, 0, stream>>>(ff_w, ffwT);
  }
  cvt_transpose_kernel<<<(C_ * 2 * DI_ + 255) / 256, 256, 0, stream>>>(w_in, winT, C_, 2 * DI_);
  cvt_transpose_kernel<<<(DI_ * 64 + 255) / 256, 256, 0, stream>>>(w_xproj, wxpT, DI_, 64);
  cvt_transpose_kernel<<<(R_ * DI_ + 255) / 256, 256, 0, stream>>>(w_dt, wdtT, R_, DI_);
  cvt_transpose_kernel<<<(DI_ * C_ + 255) / 256, 256, 0, stream>>>(w_out, woutT, DI_, C_);

  // 1) ff conv + bias + ReLU : 2048 waves (16x64 tiles)
  conv_ffn_wmma_kernel<<<256, 256, 0, stream>>>(ffwT, xpadT, ff_b, convout);
  // 2) batchnorm stats + 3) normalize/transpose
  bn_stats_kernel<<<C_, 256, 0, stream>>>(convout, bn_mean, bn_rstd);
  bn_norm_transpose_kernel<<<(B_ * C_ * L_) / 256, 256, 0, stream>>>(
      convout, bn_mean, bn_rstd, gamma, beta, mask, ubT);
  // 4) xz = u @ w_in : (4096x512)x(512x2048), blocks of 128x64 -> 1024 blocks
  gemm_bf16_wmma_kernel<<<1024, 256, 0, stream>>>(ubT, winT, xz, B_ * L_, 2 * DI_, C_);
  // 5) depthwise causal conv + SiLU
  dwconv_silu_kernel<<<(B_ * L_ * DI_) / 256, 256, 0, stream>>>(xz, conv_w, conv_b,
                                                                xs_act, xs_actb);
  // 6) dbc = xs @ w_xproj : (4096x1024)x(1024x64) -> 32 blocks
  gemm_bf16_wmma_kernel<<<32, 256, 0, stream>>>(xs_actb, wxpT, dbc, B_ * L_, 64, DI_);
  // 7) dt = softplus(dbc[:, :32] @ w_dt + bias) -> 4096 waves
  gemm_dt_wmma_kernel<<<512, 256, 0, stream>>>(dbc, wdtT, dt_bias, dtb);
  // 8) selective scan (fused y, D-skip, SiLU(z) gate)
  scan_kernel<<<8, 256, 0, stream>>>(dtb, xs_act, dbc, xz, A_log, D_skip, ygb);
  // 9) att = y @ w_out : (4096x1024)x(1024x512) -> 256 blocks
  gemm_bf16_wmma_kernel<<<256, 256, 0, stream>>>(ygb, woutT, attT, B_ * L_, C_, DI_);
  // 10) residual + mask, back to (B, C, L)
  epilogue_kernel<<<(B_ * C_ * L_) / 256, 256, 0, stream>>>(x, convout, attT, mask, out);
}
